// CTDG_Encoder_63024350102033
// MI455X (gfx1250) — matrix operand order repaired and checked by reading, not compiled, not run on hardware
//
#include <hip/hip_runtime.h>

typedef __attribute__((ext_vector_type(16))) _Float16 v16h;
typedef __attribute__((ext_vector_type(8)))  _Float16 v8h;
typedef __attribute__((ext_vector_type(8)))  float    v8f;

#define N_NODES 200000
#define DD      128
#define EE      100000
#define LAMB    30.0f
#define OUTP    30.0f
#define SLOPE   0.01f
#define EPSV    1e-10f

#define WAVES      4
#define TILES      5
#define BLOCK_ROWS (WAVES * TILES * 16)      // 320
#define GRID_MAIN  (N_NODES / BLOCK_ROWS)    // 625 (exact)

__device__ __forceinline__ float leaky(float x) { return x >= 0.0f ? x : SLOPE * x; }

// ---------------- inverse-map kernels ----------------
__global__ void inv_init(int* __restrict__ inv) {
    int i = blockIdx.x * blockDim.x + threadIdx.x;
    if (i < N_NODES) inv[i] = -1;
}

__global__ void inv_scatter(const int* __restrict__ src, int* __restrict__ inv) {
    int i = blockIdx.x * blockDim.x + threadIdx.x;
    if (i < EE) inv[src[i]] = i;   // unique sources -> no conflicts
}

// ---------------- fused main kernel ----------------
__launch_bounds__(WAVES * 32)
__global__ void ctdg_main(const float* __restrict__ memory,
                          const float* __restrict__ last_update,
                          const float* __restrict__ messages,
                          const float* __restrict__ timestamps,
                          const float* __restrict__ static_emb,
                          const float* __restrict__ W1,
                          const float* __restrict__ b1,
                          const float* __restrict__ W2,
                          const float* __restrict__ b2,
                          const float* __restrict__ e_lamb_p,
                          const float* __restrict__ now_p,
                          const int*   __restrict__ inv,
                          float*       __restrict__ out)
{
    // transposed f16 weights: Wt[n][k]
    __shared__ __align__(16) _Float16 W1t[128][256];           // 64 KB
    __shared__ __align__(16) _Float16 W2t[128][128];           // 32 KB
    __shared__ __align__(16) _Float16 h1s[WAVES][16][128];     // 16 KB (per-wave staging)
    __shared__ float b1s[128], b2s[128];
    __shared__ float luS[WAVES][16];

    const int tid = threadIdx.x;

    // ---- stage weights (f32 -> f16, transpose) ----
    for (int idx = tid; idx < 256 * 128; idx += WAVES * 32) {
        int k = idx >> 7, n = idx & 127;
        W1t[n][k] = (_Float16)W1[idx];
    }
    for (int idx = tid; idx < 128 * 128; idx += WAVES * 32) {
        int k = idx >> 7, n = idx & 127;
        W2t[n][k] = (_Float16)W2[idx];
    }
    if (tid < 128) { b1s[tid] = b1[tid]; b2s[tid] = b2[tid]; }
    __syncthreads();

    const float e_lamb = e_lamb_p[0];
    const float now_t  = now_p[0];
    const int   w    = tid >> 5;
    const int   lane = tid & 31;
    const int   m    = lane & 15;    // row-in-tile for A, col-in-tile for B/C/D
    const int   hi   = lane >> 4;

    for (int t = 0; t < TILES; ++t) {
        const int r0  = blockIdx.x * BLOCK_ROWS + (t * WAVES + w) * 16;
        const int row = r0 + m;

        // ---- fused sparse update for this lane's row ----
        const int e = inv[row];
        float decay, lu;
        const float* mrow = memory + (size_t)row * (DD + 1);
        const float* grow = nullptr;
        if (e >= 0) {
            float ts = timestamps[e];
            decay = __expf((last_update[row] - ts) * (1.0f / LAMB));
            lu = ts;
            grow = messages + (size_t)e * (DD + 1);
        } else {
            decay = 1.0f;
            lu = last_update[row];
        }
        float cnt   = mrow[DD] * decay + (grow ? grow[DD] : 0.0f);
        float denom = 1.0f / (cnt + EPSV);
        if (hi == 0) luS[w][m] = lu;

        // ---- build A fragments for GEMM1 (feat, K=256) ----
        // A fragment, this code's k order (matches B): lane m, 16 contiguous K at 32s+16hi
        // feat[k] (k<128) = row[k]*denom ; feat[k+128] = row[k]  -> share loads
        v16h a1[8];
        #pragma unroll
        for (int s = 0; s < 4; ++s) {
            #pragma unroll
            for (int c = 0; c < 2; ++c) {
                int kk = 32 * s + 16 * hi + 8 * c;   // in [0,128)
                #pragma unroll
                for (int q = 0; q < 8; ++q) {
                    float v = mrow[kk + q] * decay + (grow ? grow[kk + q] : 0.0f);
                    a1[s + 4][c * 8 + q] = (_Float16)v;            // feat high half (msum)
                    a1[s][c * 8 + q]     = (_Float16)(v * denom);  // feat low half (msum/cnt)
                }
            }
        }

        // ---- GEMM1: h1 = leaky(feat @ W1 + b1), 16x128 ----
        #pragma unroll
        for (int nt = 0; nt < 8; ++nt) {
            v8f acc = {};
            #pragma unroll
            for (int s = 0; s < 8; ++s) {
                // B fragment, same k order as A: lane col m, 16 contiguous K at 32s + 16hi
                const _Float16* bp = &W1t[nt * 16 + m][32 * s + 16 * hi];
                v8h b0 = *(const v8h*)bp;
                v8h b1v = *(const v8h*)(bp + 8);
                v16h bf;
                #pragma unroll
                for (int q = 0; q < 8; ++q) { bf[q] = b0[q]; bf[8 + q] = b1v[q]; }
                acc = __builtin_amdgcn_wmma_f32_16x16x32_f16(
                        false, a1[s], false, bf, (short)0, acc, false, false);
            }
            float bv = b1s[nt * 16 + m];
            #pragma unroll
            for (int j = 0; j < 8; ++j) {           // C/D: VGPR j -> M = j + 8*hi
                int M = j + 8 * hi;
                h1s[w][M][nt * 16 + m] = (_Float16)leaky(acc[j] + bv);
            }
        }
        // same-wave DS ordering guarantees h1s stores complete before loads below

        // ---- re-fragment h1 as A for GEMM2 (K=128) ----
        v16h a2[4];
        #pragma unroll
        for (int s = 0; s < 4; ++s) {
            const _Float16* ap = &h1s[w][m][32 * s + 16 * hi];
            v8h c0 = *(const v8h*)ap;
            v8h c1 = *(const v8h*)(ap + 8);
            #pragma unroll
            for (int q = 0; q < 8; ++q) { a2[s][q] = c0[q]; a2[s][8 + q] = c1[q]; }
        }

        // ---- GEMM2 + fused epilogue ----
        #pragma unroll
        for (int nt = 0; nt < 8; ++nt) {
            v8f acc = {};
            #pragma unroll
            for (int s = 0; s < 4; ++s) {
                const _Float16* bp = &W2t[nt * 16 + m][32 * s + 16 * hi];
                v8h b0 = *(const v8h*)bp;
                v8h b1v = *(const v8h*)(bp + 8);
                v16h bf;
                #pragma unroll
                for (int q = 0; q < 8; ++q) { bf[q] = b0[q]; bf[8 + q] = b1v[q]; }
                acc = __builtin_amdgcn_wmma_f32_16x16x32_f16(
                        false, a2[s], false, bf, (short)0, acc, false, false);
            }
            int n = nt * 16 + m;
            float bv = b2s[n];
            #pragma unroll
            for (int j = 0; j < 8; ++j) {
                int M = j + 8 * hi;
                int r = r0 + M;
                float ed = __expf((luS[w][M] - now_t) * (1.0f / OUTP));
                float h  = leaky(acc[j] + bv);
                out[(size_t)r * DD + n] =
                    e_lamb * static_emb[(size_t)r * DD + n] + (1.0f - e_lamb) * h * ed;
            }
        }
    }
}

extern "C" void kernel_launch(void* const* d_in, const int* in_sizes, int n_in,
                              void* d_out, int out_size, void* d_ws, size_t ws_size,
                              hipStream_t stream) {
    const float* memory      = (const float*)d_in[0];
    const float* last_update = (const float*)d_in[1];
    const float* messages    = (const float*)d_in[2];
    const float* timestamps  = (const float*)d_in[3];
    const float* static_emb  = (const float*)d_in[4];
    const float* W1          = (const float*)d_in[5];
    const float* b1          = (const float*)d_in[6];
    const float* W2          = (const float*)d_in[7];
    const float* b2          = (const float*)d_in[8];
    const float* e_lamb      = (const float*)d_in[9];
    const float* now_t       = (const float*)d_in[10];
    const int*   src         = (const int*)d_in[11];

    int*   inv = (int*)d_ws;            // N ints = 800 KB scratch
    float* out = (float*)d_out;

    inv_init<<<(N_NODES + 255) / 256, 256, 0, stream>>>(inv);
    inv_scatter<<<(EE + 255) / 256, 256, 0, stream>>>(src, inv);
    ctdg_main<<<GRID_MAIN, WAVES * 32, 0, stream>>>(
        memory, last_update, messages, timestamps, static_emb,
        W1, b1, W2, b2, e_lamb, now_t, inv, out);
}
